// PointNet2Seg_39316130627922
// MI455X (gfx1250) — matrix operand order repaired and checked by reading, not compiled, not run on hardware
//
#include <hip/hip_runtime.h>
#include <stdint.h>

typedef _Float16 f16;
typedef _Float16 v16h __attribute__((ext_vector_type(16)));
typedef float    v8f  __attribute__((ext_vector_type(8)));

#define BN_EPS 1e-5f

// ---------------------------------------------------------------------------
// Farthest point sampling: one block per batch element, distance array in LDS.
// ---------------------------------------------------------------------------
__global__ void k_fps(const float* __restrict__ xyz, int* __restrict__ out, int N, int S) {
  __shared__ float dist[8192];
  __shared__ float rmx[256];
  __shared__ int   ridx[256];
  __shared__ int   cur;
  const int b = blockIdx.x, t = threadIdx.x, nt = blockDim.x;
  const float* x = xyz + (size_t)b * N * 3;
  for (int i = t; i < N; i += nt) dist[i] = 1e10f;
  if (t == 0) cur = 0;
  __syncthreads();
  for (int s = 0; s < S; ++s) {
    const int f = cur;
    if (t == 0) out[b * S + s] = f;
    const float cx = x[f*3+0], cy = x[f*3+1], cz = x[f*3+2];
    float bm = -1.0f; int bi = 0;
    for (int i = t; i < N; i += nt) {
      float dx = x[i*3+0]-cx, dy = x[i*3+1]-cy, dz = x[i*3+2]-cz;
      float d  = dx*dx + dy*dy + dz*dz;
      float dd = fminf(dist[i], d);
      dist[i] = dd;
      if (dd > bm) { bm = dd; bi = i; }   // strict > keeps first occurrence (argmax tie rule)
    }
    rmx[t] = bm; ridx[t] = bi;
    __syncthreads();
    for (int w = nt >> 1; w > 0; w >>= 1) {
      if (t < w) {
        if (rmx[t+w] > rmx[t] || (rmx[t+w] == rmx[t] && ridx[t+w] < ridx[t])) {
          rmx[t] = rmx[t+w]; ridx[t] = ridx[t+w];
        }
      }
      __syncthreads();
    }
    if (t == 0) cur = ridx[0];
    __syncthreads();
  }
}

// Gather 3-vectors by index: dst[b,s,:] = src[b,idx[b,s],:]
__global__ void k_gather3(const float* __restrict__ src, const int* __restrict__ idx,
                          float* __restrict__ dst, int N, int S, int B) {
  int t = blockIdx.x*blockDim.x + threadIdx.x;
  if (t >= B*S) return;
  int b = t / S;
  int gi = idx[t];
  const float* p = src + ((size_t)b*N + gi)*3;
  dst[t*3+0] = p[0]; dst[t*3+1] = p[1]; dst[t*3+2] = p[2];
}

// Ball query: first `ns` indices (in index order) with dist^2 <= r2; pad with first hit.
__global__ void k_ballq(const float* __restrict__ xyz, const float* __restrict__ nxyz,
                        int* __restrict__ idx, int N, int S, int ns, float r2, int B) {
  int t = blockIdx.x*blockDim.x + threadIdx.x;
  if (t >= B*S) return;
  int b = t / S;
  const float qx = nxyz[t*3], qy = nxyz[t*3+1], qz = nxyz[t*3+2];
  const float* x = xyz + (size_t)b*N*3;
  int* o = idx + (size_t)t*ns;
  int cnt = 0, first = -1;
  for (int i = 0; i < N; ++i) {
    float dx = x[i*3]-qx, dy = x[i*3+1]-qy, dz = x[i*3+2]-qz;
    if (dx*dx + dy*dy + dz*dz <= r2) {
      if (first < 0) first = i;
      if (cnt < ns) o[cnt++] = i; else break;
    }
  }
  if (first < 0) first = 0;
  while (cnt < ns) o[cnt++] = first;
}

// Build grouped f16 rows: [xyz - center, gathered features], zero-padded to Cpad.
__global__ void k_group(const float* __restrict__ xyz, const float* __restrict__ nxyz,
                        const int* __restrict__ idx, const float* __restrict__ feat, int Cf,
                        int N, int S, int ns, int Cpad, f16* __restrict__ X, int B, int R) {
  int t = blockIdx.x*blockDim.x + threadIdx.x;
  if (t >= R) return;
  int bs = t / ns;
  int b  = bs / S;
  int gi = idx[t];
  const float* p = xyz + ((size_t)b*N + gi)*3;
  const float* q = nxyz + (size_t)bs*3;
  f16* o = X + (size_t)t*Cpad;
  o[0] = (f16)(p[0]-q[0]); o[1] = (f16)(p[1]-q[1]); o[2] = (f16)(p[2]-q[2]);
  int c = 3;
  if (feat) {
    const float* fr = feat + ((size_t)b*N + gi)*Cf;
    for (int k = 0; k < Cf; ++k) o[c++] = (f16)fr[k];
  }
  for (; c < Cpad; ++c) o[c] = (f16)0.0f;
}

// 3-NN inverse-distance interpolation + concat [points1, interp] into f16 rows.
__global__ void k_interp(const float* __restrict__ x1, int N1, const float* __restrict__ x2, int N2,
                         const float* __restrict__ p1, int C1, const float* __restrict__ p2, int C2,
                         int Cpad, f16* __restrict__ X, int B) {
  int t = blockIdx.x*blockDim.x + threadIdx.x;
  if (t >= B*N1) return;
  int b = t / N1;
  const float qx = x1[t*3], qy = x1[t*3+1], qz = x1[t*3+2];
  const float* xx = x2 + (size_t)b*N2*3;
  float d0 = 1e30f, d1 = 1e30f, d2v = 1e30f; int i0 = 0, i1 = 0, i2 = 0;
  for (int i = 0; i < N2; ++i) {
    float dx = xx[i*3]-qx, dy = xx[i*3+1]-qy, dz = xx[i*3+2]-qz;
    float d = dx*dx + dy*dy + dz*dz;
    if (d < d0)      { d2v = d1; i2 = i1; d1 = d0; i1 = i0; d0 = d; i0 = i; }
    else if (d < d1) { d2v = d1; i2 = i1; d1 = d;  i1 = i; }
    else if (d < d2v){ d2v = d;  i2 = i; }
  }
  float r0 = 1.f/(d0+1e-8f), r1 = 1.f/(d1+1e-8f), r2 = 1.f/(d2v+1e-8f);
  float rs = r0 + r1 + r2; r0 /= rs; r1 /= rs; r2 /= rs;
  f16* o = X + (size_t)t*Cpad;
  const float* pr = p1 + (size_t)t*C1;
  int c = 0;
  for (int k = 0; k < C1; ++k) o[c++] = (f16)pr[k];
  const float* f0 = p2 + ((size_t)b*N2+i0)*C2;
  const float* f1 = p2 + ((size_t)b*N2+i1)*C2;
  const float* f2 = p2 + ((size_t)b*N2+i2)*C2;
  for (int k = 0; k < C2; ++k) o[c++] = (f16)(r0*f0[k] + r1*f1[k] + r2*f2[k]);
  for (; c < Cpad; ++c) o[c] = (f16)0.0f;
}

// Convert f32 weights W[O][C] -> f16 W[O][Cpad] (zero-padded K).
__global__ void k_wconv(const float* __restrict__ w, f16* __restrict__ wh, int C, int Cpad, int tot) {
  int t = blockIdx.x*blockDim.x + threadIdx.x;
  if (t >= tot) return;
  int o = t / Cpad, c = t % Cpad;
  wh[t] = (c < C) ? (f16)w[(size_t)o*C + c] : (f16)0.0f;
}

// ---------------------------------------------------------------------------
// WMMA GEMM: one wave per 16x16 output tile.  Y = X(f16,[MxCpad]) * W^T + bias
// A-frag layout (16-bit A 16x32): lanes 0-15 rows M=0..15, K={0..7,16..23};
// lanes 16-31 same rows, K={8..15,24..31}.  B from W[o][c] row-major: lane n
// holds column n, K pairs contiguous -> dword loads.
// ---------------------------------------------------------------------------
__global__ __launch_bounds__(32)
void k_wmma_gemm(const f16* __restrict__ X, const f16* __restrict__ W,
                 const float* __restrict__ bias, float* __restrict__ Y, int Cpad, int O) {
  const int lane = threadIdx.x;
  const int m0 = blockIdx.x * 16, n0 = blockIdx.y * 16;
  const int r = lane & 15, hi = lane >> 4;
  const uint32_t* xrow = (const uint32_t*)(X + (size_t)(m0 + r) * Cpad);
  const uint32_t* wrow = (const uint32_t*)(W + (size_t)(n0 + r) * Cpad);
  v8f acc = {0.f,0.f,0.f,0.f,0.f,0.f,0.f,0.f};
  for (int kb = 0; kb < Cpad; kb += 32) {
    union { v16h v; uint32_t u[8]; } A, Bm;
    const uint32_t* xp = xrow + (kb >> 1);
    const uint32_t* wp = wrow + (kb >> 1);
#pragma unroll
    for (int k = 0; k < 4; ++k) { A.u[k] = xp[hi*4 + k]; A.u[4+k] = xp[8 + hi*4 + k]; }
#pragma unroll
    for (int k = 0; k < 4; ++k) { Bm.u[k] = wp[hi*4 + k]; Bm.u[4+k] = wp[8 + hi*4 + k]; }
    acc = __builtin_amdgcn_wmma_f32_16x16x32_f16(false, A.v, false, Bm.v,
                                                 (short)0, acc, false, false);
  }
  const float bv = bias[n0 + r];
  float* yb = Y + (size_t)(m0 + hi*8) * O + n0 + r;
#pragma unroll
  for (int j = 0; j < 8; ++j) yb[(size_t)j * O] = acc[j] + bv;
}

// Per-channel batch statistics: one block per channel.
__global__ void k_bn_stats(const float* __restrict__ Y, float* __restrict__ mean,
                           float* __restrict__ var, int M, int O) {
  __shared__ float s1[256], s2[256];
  int c = blockIdx.x, t = threadIdx.x;
  float a = 0.f, b = 0.f;
  for (int m = t; m < M; m += blockDim.x) {
    float v = Y[(size_t)m*O + c]; a += v; b += v*v;
  }
  s1[t] = a; s2[t] = b; __syncthreads();
  for (int w = 128; w > 0; w >>= 1) {
    if (t < w) { s1[t] += s1[t+w]; s2[t] += s2[t+w]; }
    __syncthreads();
  }
  if (t == 0) { float mu = s1[0]/(float)M; mean[c] = mu; var[c] = s2[0]/(float)M - mu*mu; }
}

// Normalize + scale + ReLU; write f32 (possibly in-place) and optional f16 for next GEMM.
__global__ void k_bn_relu(const float* Y, float* Yo, const float* __restrict__ mean,
                          const float* __restrict__ var, const float* __restrict__ g,
                          const float* __restrict__ be, int O, int tot, f16* Xn) {
  int t = blockIdx.x*blockDim.x + threadIdx.x;
  if (t >= tot) return;
  int c = t % O;
  float a = g[c] * (Y[t] - mean[c]) * rsqrtf(var[c] + BN_EPS) + be[c];
  a = fmaxf(a, 0.0f);
  Yo[t] = a;
  if (Xn) Xn[t] = (f16)a;
}

// Max over the nsample axis.
__global__ void k_maxpool(const float* __restrict__ Y, float* __restrict__ out,
                          int ns, int O, int BS) {
  int t = blockIdx.x*blockDim.x + threadIdx.x;
  if (t >= BS*O) return;
  int bs = t / O, c = t % O;
  float m = -1e30f;
  const float* p = Y + (size_t)bs*ns*O + c;
  for (int j = 0; j < ns; ++j) m = fmaxf(m, p[(size_t)j*O]);
  out[t] = m;
}

// Final 128->6 head with output transpose to (B, 6, N).
__global__ void k_seg2(const float* __restrict__ X, const float* __restrict__ w,
                       const float* __restrict__ b, float* __restrict__ out, int N, int B) {
  int t = blockIdx.x*blockDim.x + threadIdx.x;
  if (t >= B*N) return;
  int bb = t / N, n = t % N;
  const float* x = X + (size_t)t * 128;
  for (int o = 0; o < 6; ++o) {
    float acc = b[o];
    const float* wr = w + o*128;
    for (int c = 0; c < 128; ++c) acc += x[c]*wr[c];
    out[((size_t)bb*6 + o)*N + n] = acc;
  }
}

// ---------------------------------------------------------------------------
// Host orchestration
// ---------------------------------------------------------------------------
struct LP { const float *w, *b, *g, *be; int C, O; };

static void run_layer(hipStream_t st, const f16* X, int M, int Cpad, const LP& lp,
                      f16* Wh, float* Y, float* Yout, float* mean, float* var, f16* Xnext) {
  const int O = lp.O;
  int wcnt = O * Cpad;
  k_wconv<<<(wcnt + 255)/256, 256, 0, st>>>(lp.w, Wh, lp.C, Cpad, wcnt);
  dim3 g((unsigned)(M/16), (unsigned)(O/16));
  k_wmma_gemm<<<g, 32, 0, st>>>(X, Wh, lp.b, Y, Cpad, O);
  k_bn_stats<<<O, 256, 0, st>>>(Y, mean, var, M, O);
  int tot = M * O;
  k_bn_relu<<<(tot + 255)/256, 256, 0, st>>>(Y, Yout, mean, var, lp.g, lp.be, O, tot, Xnext);
}

extern "C" void kernel_launch(void* const* d_in, const int* in_sizes, int n_in,
                              void* d_out, int out_size, void* d_ws, size_t ws_size,
                              hipStream_t stream) {
  const int B = 16, N0 = 8192;
  const float* xyz = (const float*)d_in[0];
  float* out = (float*)d_out;

  // --- parameter pointer extraction -----------------------------------------
  // Hypothesis A (JAX pytree flatten, dict keys sorted): params order
  //   fp1,fp2,fp3,sa1,sa2,sa3,seg1,seg2 ; layer leaves b,beta,gamma,w ; seg2 b,w.
  // Hypothesis B (insertion order): sa1,sa2,sa3,fp3,fp2,fp1,seg1,seg2 ;
  //   layer leaves w,b,gamma,beta ; seg2 w,b.
  bool alpha = (n_in > 4) ? (in_sizes[4] > 1000) : true;  // A: idx4 = fp1.w0 (128*131)

  LP sa1[3], sa2[3], sa3[3], fp3[2], fp2[2], fp1[3], seg1;
  const float *seg2w = nullptr, *seg2b = nullptr;
  auto grab = [&](int base, int C, int O) -> LP {
    LP l; l.C = C; l.O = O;
    if (alpha) {
      l.b  = (const float*)d_in[base+0]; l.be = (const float*)d_in[base+1];
      l.g  = (const float*)d_in[base+2]; l.w  = (const float*)d_in[base+3];
    } else {
      l.w  = (const float*)d_in[base+0]; l.b  = (const float*)d_in[base+1];
      l.g  = (const float*)d_in[base+2]; l.be = (const float*)d_in[base+3];
    }
    return l;
  };
  auto fill = [&](LP* dst, int nl, int cin, const int* outs, int& cur) {
    int c = cin;
    for (int j = 0; j < nl; ++j) { dst[j] = grab(cur, c, outs[j]); cur += 4; c = outs[j]; }
  };
  const int o_sa1[3] = {32,64,128}, o_sa2[3] = {128,128,256}, o_sa3[3] = {256,512,1024};
  const int o_fp3[2] = {256,256},   o_fp2[2] = {256,128},     o_fp1[3] = {128,128,128};
  const int o_seg1[1] = {128};
  int cur = 1;
  if (alpha) {
    fill(fp1, 3, 131, o_fp1, cur);
    fill(fp2, 2, 384, o_fp2, cur);
    fill(fp3, 2, 1280, o_fp3, cur);
    fill(sa1, 3, 3,   o_sa1, cur);
    fill(sa2, 3, 131, o_sa2, cur);
    fill(sa3, 3, 259, o_sa3, cur);
    fill(&seg1, 1, 128, o_seg1, cur);
    seg2b = (const float*)d_in[cur+0]; seg2w = (const float*)d_in[cur+1];
  } else {
    fill(sa1, 3, 3,   o_sa1, cur);
    fill(sa2, 3, 131, o_sa2, cur);
    fill(sa3, 3, 259, o_sa3, cur);
    fill(fp3, 2, 1280, o_fp3, cur);
    fill(fp2, 2, 384, o_fp2, cur);
    fill(fp1, 3, 131, o_fp1, cur);
    fill(&seg1, 1, 128, o_seg1, cur);
    seg2w = (const float*)d_in[cur+0]; seg2b = (const float*)d_in[cur+1];
  }

  // --- workspace carve-up ----------------------------------------------------
  char* w8 = (char*)d_ws;
  size_t off = 0;
  auto alloc = [&](size_t bytes) -> void* {
    void* p = w8 + off; off += (bytes + 255) & ~(size_t)255; return p;
  };
  float* Y     = (float*)alloc((size_t)262144 * 128 * 4);       // 134 MB, reused every layer
  f16*   Xh    = (f16*)  alloc((size_t)48 * 1024 * 1024);       // f16 activations (max 42 MB)
  f16*   Wh    = (f16*)  alloc((size_t)2 * 1024 * 1024);        // f16 weights (max 1 MB)
  float* mean  = (float*)alloc(4096);
  float* var   = (float*)alloc(4096);
  int*   fpsi  = (int*)  alloc((size_t)B * 512 * 4);
  int*   idx   = (int*)  alloc((size_t)B * 512 * 32 * 4);
  float* l1xyz = (float*)alloc((size_t)B * 512 * 3 * 4);
  float* l2xyz = (float*)alloc((size_t)B * 128 * 3 * 4);
  float* l3xyz = (float*)alloc((size_t)B * 32 * 3 * 4);
  float* l1    = (float*)alloc((size_t)B * 512 * 128 * 4);
  float* l2    = (float*)alloc((size_t)B * 128 * 256 * 4);
  float* l3    = (float*)alloc((size_t)B * 32 * 1024 * 4);
  float* l2f   = (float*)alloc((size_t)B * 128 * 256 * 4);
  float* l1f   = (float*)alloc((size_t)B * 512 * 128 * 4);
  (void)ws_size; (void)out_size;

  hipStream_t st = stream;

  // ---- SA1: npoint=512, nsample=32, r=0.2, MLP 3->32->64->128 ---------------
  k_fps<<<B, 256, 0, st>>>(xyz, fpsi, N0, 512);
  k_gather3<<<(B*512 + 255)/256, 256, 0, st>>>(xyz, fpsi, l1xyz, N0, 512, B);
  k_ballq<<<(B*512 + 255)/256, 256, 0, st>>>(xyz, l1xyz, idx, N0, 512, 32, 0.04f, B);
  int M1 = B * 512 * 32;  // 262144
  k_group<<<(M1 + 255)/256, 256, 0, st>>>(xyz, l1xyz, idx, nullptr, 0, N0, 512, 32, 32, Xh, B, M1);
  run_layer(st, Xh, M1, 32, sa1[0], Wh, Y, Y, mean, var, Xh);
  run_layer(st, Xh, M1, 32, sa1[1], Wh, Y, Y, mean, var, Xh);
  run_layer(st, Xh, M1, 64, sa1[2], Wh, Y, Y, mean, var, nullptr);
  k_maxpool<<<(B*512*128 + 255)/256, 256, 0, st>>>(Y, l1, 32, 128, B*512);

  // ---- SA2: npoint=128, nsample=64, r=0.4, MLP 131->128->128->256 -----------
  k_fps<<<B, 256, 0, st>>>(l1xyz, fpsi, 512, 128);
  k_gather3<<<(B*128 + 255)/256, 256, 0, st>>>(l1xyz, fpsi, l2xyz, 512, 128, B);
  k_ballq<<<(B*128 + 255)/256, 256, 0, st>>>(l1xyz, l2xyz, idx, 512, 128, 64, 0.16f, B);
  int M2 = B * 128 * 64;  // 131072
  k_group<<<(M2 + 255)/256, 256, 0, st>>>(l1xyz, l2xyz, idx, l1, 128, 512, 128, 64, 160, Xh, B, M2);
  run_layer(st, Xh, M2, 160, sa2[0], Wh, Y, Y, mean, var, Xh);
  run_layer(st, Xh, M2, 128, sa2[1], Wh, Y, Y, mean, var, Xh);
  run_layer(st, Xh, M2, 128, sa2[2], Wh, Y, Y, mean, var, nullptr);
  k_maxpool<<<(B*128*256 + 255)/256, 256, 0, st>>>(Y, l2, 64, 256, B*128);

  // ---- SA3: npoint=32, nsample=64, r=0.8, MLP 259->256->512->1024 -----------
  k_fps<<<B, 256, 0, st>>>(l2xyz, fpsi, 128, 32);
  k_gather3<<<(B*32 + 255)/256, 256, 0, st>>>(l2xyz, fpsi, l3xyz, 128, 32, B);
  k_ballq<<<(B*32 + 255)/256, 256, 0, st>>>(l2xyz, l3xyz, idx, 128, 32, 64, 0.64f, B);
  int M3 = B * 32 * 64;   // 32768
  k_group<<<(M3 + 255)/256, 256, 0, st>>>(l2xyz, l3xyz, idx, l2, 256, 128, 32, 64, 288, Xh, B, M3);
  run_layer(st, Xh, M3, 288, sa3[0], Wh, Y, Y, mean, var, Xh);
  run_layer(st, Xh, M3, 256, sa3[1], Wh, Y, Y, mean, var, Xh);
  run_layer(st, Xh, M3, 512, sa3[2], Wh, Y, Y, mean, var, nullptr);
  k_maxpool<<<(B*32*1024 + 255)/256, 256, 0, st>>>(Y, l3, 64, 1024, B*32);

  // ---- FP3: interp l3(1024) onto l2_xyz, concat l2(256) -> MLP 1280->256->256
  int Mf3 = B * 128;
  k_interp<<<(Mf3 + 255)/256, 256, 0, st>>>(l2xyz, 128, l3xyz, 32, l2, 256, l3, 1024, 1280, Xh, B);
  run_layer(st, Xh, Mf3, 1280, fp3[0], Wh, Y, Y, mean, var, Xh);
  run_layer(st, Xh, Mf3, 256,  fp3[1], Wh, Y, l2f, mean, var, nullptr);

  // ---- FP2: interp l2f(256) onto l1_xyz, concat l1(128) -> MLP 384->256->128
  int Mf2 = B * 512;
  k_interp<<<(Mf2 + 255)/256, 256, 0, st>>>(l1xyz, 512, l2xyz, 128, l1, 128, l2f, 256, 384, Xh, B);
  run_layer(st, Xh, Mf2, 384, fp2[0], Wh, Y, Y, mean, var, Xh);
  run_layer(st, Xh, Mf2, 256, fp2[1], Wh, Y, l1f, mean, var, nullptr);

  // ---- FP1: interp l1f(128) onto xyz, concat xyz(3) -> MLP 131->128->128->128
  int Mf1 = B * N0;       // 131072
  k_interp<<<(Mf1 + 255)/256, 256, 0, st>>>(xyz, N0, l1xyz, 512, xyz, 3, l1f, 128, 160, Xh, B);
  run_layer(st, Xh, Mf1, 160, fp1[0], Wh, Y, Y, mean, var, Xh);
  run_layer(st, Xh, Mf1, 128, fp1[1], Wh, Y, Y, mean, var, Xh);
  run_layer(st, Xh, Mf1, 128, fp1[2], Wh, Y, Y, mean, var, Xh);

  // ---- seg1 (128->128, BN+ReLU) then seg2 (128->6) with transpose -----------
  run_layer(st, Xh, Mf1, 128, seg1, Wh, Y, Y, mean, var, nullptr);
  k_seg2<<<(Mf1 + 255)/256, 256, 0, st>>>(Y, seg2w, seg2b, out, N0, B);
}
